// AttentionLayer_89472758710963
// MI455X (gfx1250) — matrix operand order repaired and checked by reading, not compiled, not run on hardware
//
#include <hip/hip_runtime.h>

typedef __bf16 bf16_t;
typedef __attribute__((ext_vector_type(16))) __bf16        v16bf;
typedef __attribute__((ext_vector_type(8)))  float         v8f;
typedef __attribute__((ext_vector_type(4)))  unsigned int  u32x4;
typedef __attribute__((ext_vector_type(4)))  float         f32x4;
typedef __attribute__((ext_vector_type(4)))  __bf16        bf16x4;

static constexpr int BATCH = 32;
static constexpr int SEQ   = 1024;
static constexpr int DIM   = 1024;

union FragBF { v16bf v; u32x4 q[2]; };

// ---------------- fp32 -> bf16 elementwise ----------------
__global__ void cvt_f32_bf16(const float* __restrict__ in, bf16_t* __restrict__ out, size_t n)
{
    size_t i = ((size_t)blockIdx.x * blockDim.x + threadIdx.x) * 4;
    if (i >= n) return;
    f32x4 v = *(const f32x4*)(in + i);
    bf16x4 o;
    o.x = (bf16_t)v.x; o.y = (bf16_t)v.y; o.z = (bf16_t)v.z; o.w = (bf16_t)v.w;
    *(bf16x4*)(out + i) = o;
}

// ---------------- (B,S,D) fp32 -> (B,D,S) bf16 tiled transpose ----------------
__global__ __launch_bounds__(256)
void transpose_f32_bf16(const float* __restrict__ in, bf16_t* __restrict__ out)
{
    __shared__ bf16_t tile[32][33];
    const int b  = blockIdx.z;
    const int d0 = blockIdx.x * 32;
    const int s0 = blockIdx.y * 32;
    const float* inb  = in  + (size_t)b * SEQ * DIM;
    bf16_t*      outb = out + (size_t)b * SEQ * DIM;
    const int tx = threadIdx.x, ty = threadIdx.y;  // (32, 8)
#pragma unroll
    for (int k = 0; k < 4; ++k)
        tile[ty * 4 + k][tx] = (bf16_t)inb[(size_t)(s0 + ty * 4 + k) * DIM + d0 + tx];
    __syncthreads();
#pragma unroll
    for (int k = 0; k < 4; ++k)
        outb[(size_t)(d0 + ty * 4 + k) * SEQ + s0 + tx] = tile[tx][ty * 4 + k];
}

// ---------------- fused scores -> row-softmax -> prob @ V ----------------
// Q  : (B,S,D) bf16  rows of the score matrix
// Km : (B,S,D) bf16  keys (score cols)
// Vt : (B,D,S) bf16  values, transposed so B-fragments are contiguous
// out: (B,S,D) fp32
static constexpr int ES_STRIDE = SEQ + 8;    // fp32 score-strip row stride (pad vs bank conflicts)
static constexpr int PS_STRIDE = SEQ + 16;   // bf16 prob-strip row stride

__global__ __launch_bounds__(256)
void attn_strip(const bf16_t* __restrict__ Q,
                const bf16_t* __restrict__ Km,
                const bf16_t* __restrict__ Vt,
                float* __restrict__ out)
{
    __shared__ __align__(16) float  es[16 * ES_STRIDE];   // 66 KB
    __shared__ __align__(16) bf16_t ps[16 * PS_STRIDE];   // 33 KB
    __shared__ float red[16 * 16];

    const int tid  = threadIdx.x;
    const int wave = tid >> 5;
    const int lane = tid & 31;
    const int half = lane >> 4;   // which 16-lane half of the wave
    const int ln   = lane & 15;

    const int b  = blockIdx.y;
    const int i0 = blockIdx.x * 16;

    const bf16_t* Qb = Q  + (size_t)b * SEQ * DIM;
    const bf16_t* Kb = Km + (size_t)b * SEQ * DIM;
    const bf16_t* Vb = Vt + (size_t)b * DIM * SEQ;

    const v8f vzero = {0.f, 0.f, 0.f, 0.f, 0.f, 0.f, 0.f, 0.f};

    // ===== GEMM1: e[i,j] = sum_d Q[i,d]*K[j,d], 16 rows x 1024 cols =====
    v8f acc[8];
#pragma unroll
    for (int jj = 0; jj < 8; ++jj) acc[jj] = vzero;

    const bf16_t* qrow = Qb + (size_t)(i0 + ln) * DIM;
    for (int kt = 0; kt < DIM / 32; ++kt) {
        // A fragment (16x32 bf16): lane ln row, K {0-7,16-23} / {8-15,24-31}
        FragBF a;
        const bf16_t* ab = qrow + kt * 32 + half * 8;
        a.q[0] = *(const u32x4*)(ab);
        a.q[1] = *(const u32x4*)(ab + 16);
#pragma unroll
        for (int jj = 0; jj < 8; ++jj) {
            const int j0 = (wave * 8 + jj) * 16;
            // B fragment (32x16): lane ln column = K-row j0+ln; K {0-7,16-23} / {8-15,24-31}
            const bf16_t* bb = Kb + (size_t)(j0 + ln) * DIM + kt * 32 + half * 8;
            FragBF bm;
            bm.q[0] = *(const u32x4*)(bb);
            bm.q[1] = *(const u32x4*)(bb + 16);
            acc[jj] = __builtin_amdgcn_wmma_f32_16x16x32_bf16(
                false, a.v, false, bm.v, (short)0, acc[jj], false, false);
        }
    }

    // C/D layout: vgpr r -> row r (lanes 0-15) / row r+8 (lanes 16-31), col = ln
#pragma unroll
    for (int jj = 0; jj < 8; ++jj) {
        const int j0 = (wave * 8 + jj) * 16;
#pragma unroll
        for (int r = 0; r < 8; ++r)
            es[(r + half * 8) * ES_STRIDE + j0 + ln] = acc[jj][r];
    }
    __syncthreads();

    // ===== exact row softmax over the 1024 columns, then bf16 probs =====
    {
        const int row = tid >> 4;     // 16 threads per row
        const int sub = tid & 15;
        float* erow = &es[row * ES_STRIDE];
        float m = -3.402823466e38f;
        for (int c = sub; c < SEQ; c += 16) m = fmaxf(m, erow[c]);
        red[row * 16 + sub] = m;
        __syncthreads();
        if (sub == 0) {
            float mm = red[row * 16];
#pragma unroll
            for (int t = 1; t < 16; ++t) mm = fmaxf(mm, red[row * 16 + t]);
            red[row * 16] = mm;
        }
        __syncthreads();
        m = red[row * 16];
        __syncthreads();
        float lsum = 0.f;
        for (int c = sub; c < SEQ; c += 16) {
            float v = __expf(erow[c] - m);
            erow[c] = v;
            lsum += v;
        }
        red[row * 16 + sub] = lsum;
        __syncthreads();
        if (sub == 0) {
            float s = 0.f;
#pragma unroll
            for (int t = 0; t < 16; ++t) s += red[row * 16 + t];
            red[row * 16] = s;
        }
        __syncthreads();
        const float inv = 1.0f / red[row * 16];
        for (int c = sub; c < SEQ; c += 16)
            ps[row * PS_STRIDE + c] = (bf16_t)(erow[c] * inv);
    }
    __syncthreads();

    // ===== GEMM2: out[i,d] = sum_j p[i,j] * Vt[d,j] =====
    v8f acc2[8];
#pragma unroll
    for (int jj = 0; jj < 8; ++jj) acc2[jj] = vzero;

    const bf16_t* prow = &ps[ln * PS_STRIDE];
    for (int kt = 0; kt < SEQ / 32; ++kt) {
        FragBF a;
        const bf16_t* ab = prow + kt * 32 + half * 8;   // A fragment from LDS probs
        a.q[0] = *(const u32x4*)(ab);
        a.q[1] = *(const u32x4*)(ab + 16);
#pragma unroll
        for (int jj = 0; jj < 8; ++jj) {
            const int d0 = (wave * 8 + jj) * 16;
            const bf16_t* bb = Vb + (size_t)(d0 + ln) * SEQ + kt * 32 + half * 8;
            FragBF bm;
            bm.q[0] = *(const u32x4*)(bb);
            bm.q[1] = *(const u32x4*)(bb + 16);
            acc2[jj] = __builtin_amdgcn_wmma_f32_16x16x32_bf16(
                false, a.v, false, bm.v, (short)0, acc2[jj], false, false);
        }
    }

    float* ob = out + (size_t)b * SEQ * DIM;
#pragma unroll
    for (int jj = 0; jj < 8; ++jj) {
        const int d0 = (wave * 8 + jj) * 16;
#pragma unroll
        for (int r = 0; r < 8; ++r)
            ob[(size_t)(i0 + r + half * 8) * DIM + d0 + ln] = acc2[jj][r];
    }
}

// ---------------- launch ----------------
extern "C" void kernel_launch(void* const* d_in, const int* in_sizes, int n_in,
                              void* d_out, int out_size, void* d_ws, size_t ws_size,
                              hipStream_t stream)
{
    (void)in_sizes; (void)n_in; (void)out_size; (void)ws_size;
    const float* P = (const float*)d_in[0];
    const float* H = (const float*)d_in[1];
    const size_t per = (size_t)BATCH * SEQ * DIM;

    // workspace: 4 bf16 tensors of `per` elements = 256 MB total
    bf16_t* Pbf = (bf16_t*)d_ws;
    bf16_t* Hbf = Pbf + per;
    bf16_t* Ptb = Hbf + per;   // (B,D,S)
    bf16_t* Htb = Ptb + per;   // (B,D,S)

    float* out_p = (float*)d_out;
    float* out_h = out_p + per;

    const int cvt_blocks = (int)(per / 4 / 256);
    cvt_f32_bf16<<<cvt_blocks, 256, 0, stream>>>(P, Pbf, per);
    cvt_f32_bf16<<<cvt_blocks, 256, 0, stream>>>(H, Hbf, per);

    dim3 tgrid(DIM / 32, SEQ / 32, BATCH);
    transpose_f32_bf16<<<tgrid, dim3(32, 8), 0, stream>>>(P, Ptb);
    transpose_f32_bf16<<<tgrid, dim3(32, 8), 0, stream>>>(H, Htb);

    dim3 agrid(SEQ / 16, BATCH);
    // attention_p = rowsoftmax(P·Hᵀ)·H
    attn_strip<<<agrid, 256, 0, stream>>>(Pbf, Hbf, Htb, out_p);
    // attention_h = colsoftmax(P·Hᵀ)ᵀ·P = rowsoftmax(H·Pᵀ)·P
    attn_strip<<<agrid, 256, 0, stream>>>(Hbf, Pbf, Ptb, out_h);
}
